// RNN2Layer_67044439490952
// MI455X (gfx1250) — compile-verified, matrix operand order checked by reading, not run on hardware
//
#include <hip/hip_runtime.h>
#include <math.h>

// ---------------------------------------------------------------------------
// CT-RNN Euler scan for MI455X (gfx1250, wave32, WMMA).
//   s <- s + eps*(-s + tanh(s)@Wblk + s@(A - A^T) + x_t@Win)
// T=2048, B=16, I=128, OUT=256, H=1024 (4 blocks of 256), eps=0.01.
//
// Design (round 4):
//  * 16 persistent workgroups; wg owns 64 columns (4 WMMA N-tiles) of H.
//  * Weights f32->f16, pre-swizzled into d_ws in the exact
//    v_wmma_f32_16x16x32_f16 B-operand per-lane layout (32B/lane per tile).
//  * Scan: 512 threads = 16 waves; K split 4 ways per N-tile:
//      kg0: tanh(s_j)@Wblk (8 tiles)   kg1: s_{j-1}@A[j-1,j] (8 tiles)
//      kg2: s_{j+1}@(-A^T)  (8 tiles)  kg3: x_t@Win          (4 tiles)
//    B-operands register-resident (<=64 VGPRs/wave) for all 2048 steps;
//    serial chain depth is 8 WMMAs; A operand 3-deep software-pipelined
//    (low pressure -> rotation can survive regalloc and grade the dscnt
//    waits); partials merged through a 12KB LDS buffer by kg0.
//  * embeddings[t] (d_out) doubles as the cross-WGP state exchange buffer;
//    one device-wide barrier per step (monotonic counter, agent acq/rel).
//  * couple_mask ignored: mask = (couplings != 0) so mask*couplings ==
//    couplings elementwise.
//  * Output projection: one block per M-tile; A panel staged once in LDS
//    (embeddings read exactly once), 8 waves x 2 independent WMMA chains.
// ---------------------------------------------------------------------------

#define T_STEPS 2048
#define BATCH   16
#define IN_DIM  128
#define OUT_DIM 256
#define HID     1024
#define BS      256
#define NWG     16
#define SCAN_THREADS 512

typedef __attribute__((ext_vector_type(16))) _Float16 v16h;
typedef __attribute__((ext_vector_type(8)))  float    v8f;

// ---- swizzled-weight regions in d_ws (units: _Float16 elements) ----------
#define TILE_H   512                       // halfs per [32x16] tile
#define OFF_WB   0                         // blocks diag:  [j:4][n:16][k:8]
#define SZ_WB    (4 * 16 * 8 * TILE_H)
#define OFF_WCUP (OFF_WB + SZ_WB)          // A[j-1,j]:     [j-1:3][n:16][k:8]
#define SZ_WCUP  (3 * 16 * 8 * TILE_H)
#define OFF_WCDN (OFF_WCUP + SZ_WCUP)      // -A[j,j+1]^T:  [j:3][n:16][k:8]
#define SZ_WCDN  (3 * 16 * 8 * TILE_H)
#define OFF_WIN  (OFF_WCDN + SZ_WCDN)      // input_mat:    [n:64][k:4]
#define SZ_WIN   (64 * 4 * TILE_H)
#define OFF_WOUT (OFF_WIN + SZ_WIN)        // out_mat:      [n:16][k:32]
#define SZ_WOUT  (16 * 32 * TILE_H)
#define SZ_TOTAL (OFF_WOUT + SZ_WOUT)      // 1,048,576 halfs = 2 MB
#define WS_CTR_BYTES 256

// (lane, idx) -> K within a 32-deep K-tile (16-bit operand striping)
__device__ __forceinline__ int k_of(int lane, int idx) {
  return ((idx & 8) << 1) | ((lane >> 4) << 3) | (idx & 7);
}
// K within tile -> (half, idx); lane = (half<<4) | M (A) or | N (B)
__device__ __forceinline__ void inv_k(int kk, int& half, int& idx) {
  half = (kk >> 3) & 1;
  idx  = (kk & 7) | ((kk & 16) >> 1);
}

#define WMMA(a, b, c) \
  __builtin_amdgcn_wmma_f32_16x16x32_f16(false, (a), false, (b), (short)0, (c), false, false)

// Pipelined WMMA chain over one A segment against register-resident B tiles.
// 3 rotating A registers so ds_load latency can overlap the chain.
template <int N>
__device__ __forceinline__ v8f wmma_chain(const _Float16* __restrict__ aBase,
                                          const v16h (&rB)[8], int lo, v8f acc) {
#define LDK(k) (*(const v16h*)(aBase + (k) * TILE_H + lo))
  v16h A0 = LDK(0), A1 = A0, A2 = A0;
  if (N > 1) A1 = LDK(1);
  if (N > 2) A2 = LDK(2);
#pragma unroll
  for (int k = 0; k < N; ++k) {
    v16h a = (k % 3 == 0) ? A0 : (k % 3 == 1) ? A1 : A2;
    acc = WMMA(a, rB[k], acc);
    if (k + 3 < N) {
      v16h nxt = LDK(k + 3);
      if (k % 3 == 0) A0 = nxt; else if (k % 3 == 1) A1 = nxt; else A2 = nxt;
    }
  }
#undef LDK
  return acc;
}

// ---------------------------------------------------------------------------
// Init: convert + swizzle all weights into WMMA B-operand layout in d_ws.
// ---------------------------------------------------------------------------
__global__ void __launch_bounds__(256) swizzle_weights_kernel(
    const float* __restrict__ blocks, const float* __restrict__ couplings,
    const float* __restrict__ input_mat, const float* __restrict__ out_mat,
    _Float16* __restrict__ w) {
  int e = blockIdx.x * 256 + threadIdx.x;
  if (e >= SZ_TOTAL) return;
  const int lane = (e >> 4) & 31;
  const int idx  = e & 15;
  const int kin  = k_of(lane, idx);
  const int nn   = lane & 15;
  float v;
  if (e < (int)OFF_WCUP) {                       // block-diagonal recurrent
    int r = e >> 9;
    int k = r & 7;  r >>= 3;
    int n = r & 15; r >>= 4;
    int j = r;
    v = blocks[(size_t)(j * BS + k * 32 + kin) * HID + j * BS + n * 16 + nn];
  } else if (e < (int)OFF_WCDN) {                // A[j-1,j], dest block j=1..3
    int r = (e - OFF_WCUP) >> 9;
    int k = r & 7;  r >>= 3;
    int n = r & 15; r >>= 4;
    int j = r + 1;
    v = couplings[(size_t)((j - 1) * BS + k * 32 + kin) * HID + j * BS + n * 16 + nn];
  } else if (e < (int)OFF_WIN) {                 // -A[j,j+1]^T, dest j=0..2
    int r = (e - OFF_WCDN) >> 9;
    int k = r & 7;  r >>= 3;
    int n = r & 15; r >>= 4;
    int j = r;
    v = -couplings[(size_t)(j * BS + n * 16 + nn) * HID + (j + 1) * BS + k * 32 + kin];
  } else if (e < (int)OFF_WOUT) {                // input projection
    int r = (e - OFF_WIN) >> 9;
    int k = r & 3;  r >>= 2;
    v = input_mat[(size_t)(k * 32 + kin) * HID + r * 16 + nn];
  } else {                                       // output projection
    int r = (e - OFF_WOUT) >> 9;
    int k = r & 31; r >>= 5;
    v = out_mat[(size_t)(k * 32 + kin) * OUT_DIM + r * 16 + nn];
  }
  w[e] = (_Float16)v;
}

// ---------------------------------------------------------------------------
// Persistent scan kernel: 16 workgroups x 512 threads (16 waves).
// wg = j*4 + sub owns columns [j*256 + sub*64, +64).
// Wave w: n-tile = w&3, K-group kg = w>>2 (see header).
// ---------------------------------------------------------------------------
__global__ void __launch_bounds__(SCAN_THREADS, 1) rnn_scan_kernel(
    const float* __restrict__ x, const _Float16* __restrict__ w,
    float* __restrict__ emb, unsigned* __restrict__ counter) {
  __shared__ _Float16 AsT[8 * TILE_H];  // tanh(s) own block, K=256
  __shared__ _Float16 AsU[8 * TILE_H];  // s block j-1
  __shared__ _Float16 AsD[8 * TILE_H];  // s block j+1
  __shared__ _Float16 AsX[4 * TILE_H];  // x_t, K=128
  __shared__ float    Red[3 * 1024];    // partials: [slot:3][r:8][nt:4][lane:32]

  const int tid   = threadIdx.x;
  const int lane  = tid & 31;
  const int warp  = tid >> 5;
  const int wg    = blockIdx.x;
  const int j     = wg >> 2;
  const int sub   = wg & 3;
  const int gcol0 = j * BS + sub * 64;
  const int nt    = warp & 3;           // n-tile 0..3 within this wg
  const int kg    = warp >> 2;          // K-group 0..3
  const int n     = sub * 4 + nt;       // n-tile within block j (0..15)
  const int ng    = (gcol0 >> 4) + nt;  // global n-tile for WIN (0..63)
  const int lo    = lane << 4;

  // ---- hoist B-operands into registers (invariant across all steps) ------
  v16h rB[8];
  {
    v16h z = {};
#pragma unroll
    for (int k = 0; k < 8; ++k) rB[k] = z;
  }
  if (kg == 0) {
#pragma unroll
    for (int k = 0; k < 8; ++k)
      rB[k] = *(const v16h*)&w[OFF_WB + (size_t)((j * 16 + n) * 8 + k) * TILE_H + lo];
  } else if (kg == 1) {
    if (j > 0) {
#pragma unroll
      for (int k = 0; k < 8; ++k)
        rB[k] = *(const v16h*)&w[OFF_WCUP + (size_t)(((j - 1) * 16 + n) * 8 + k) * TILE_H + lo];
    }
  } else if (kg == 2) {
    if (j < 3) {
#pragma unroll
      for (int k = 0; k < 8; ++k)
        rB[k] = *(const v16h*)&w[OFF_WCDN + (size_t)((j * 16 + n) * 8 + k) * TILE_H + lo];
    }
  } else {
#pragma unroll
    for (int k = 0; k < 4; ++k)
      rB[k] = *(const v16h*)&w[OFF_WIN + (size_t)(ng * 4 + k) * TILE_H + lo];
  }

  for (int t = 0; t < T_STEPS; ++t) {
    const float* embPrev =
        (t > 0) ? emb + (size_t)(t - 1) * (BATCH * HID) : nullptr;

    // ------- stage A operands into LDS (all 16 waves) -------
    if (t == 0) {
      for (int e = tid; e < 8 * TILE_H; e += SCAN_THREADS) {
        AsT[e] = (_Float16)0.f; AsU[e] = (_Float16)0.f; AsD[e] = (_Float16)0.f;
      }
    } else {
      for (int e = tid; e < BATCH * BS; e += SCAN_THREADS) {
        int b = e >> 8, kcol = e & 255;
        int kt = kcol >> 5, half, idx;
        inv_k(kcol & 31, half, idx);
        int dst = kt * TILE_H + (((half << 4) | b) << 4) + idx;
        float vt = embPrev[(size_t)b * HID + j * BS + kcol];
        AsT[dst] = (_Float16)tanhf(vt);
        if (j > 0) AsU[dst] = (_Float16)embPrev[(size_t)b * HID + (j - 1) * BS + kcol];
        if (j < 3) AsD[dst] = (_Float16)embPrev[(size_t)b * HID + (j + 1) * BS + kcol];
      }
    }
    for (int e = tid; e < BATCH * IN_DIM; e += SCAN_THREADS) {
      int b = e >> 7, i = e & 127;
      int kt = i >> 5, half, idx;
      inv_k(i & 31, half, idx);
      AsX[kt * TILE_H + (((half << 4) | b) << 4) + idx] =
          (_Float16)x[((size_t)t * BATCH + b) * IN_DIM + i];
    }
    // prefetch next step's input row block into the caches (gfx1250 path)
    if (warp == 15 && t + 1 < T_STEPS)
      __builtin_prefetch(x + ((size_t)(t + 1) * BATCH) * IN_DIM + lane * 64, 0, 1);
    __syncthreads();

    // ------- WMMA: register-resident B, pipelined LDS A; depth <= 8 -------
    v8f acc = {};
    if (kg == 0) {
      acc = wmma_chain<8>(AsT, rB, lo, acc);           // merged later
    } else if (kg == 1) {
      if (j > 0) {
        acc = wmma_chain<8>(AsU, rB, lo, acc);
#pragma unroll
        for (int r = 0; r < 8; ++r)
          Red[0 * 1024 + r * 128 + nt * 32 + lane] = acc[r];
      }
    } else if (kg == 2) {
      if (j < 3) {
        acc = wmma_chain<8>(AsD, rB, lo, acc);
#pragma unroll
        for (int r = 0; r < 8; ++r)
          Red[1 * 1024 + r * 128 + nt * 32 + lane] = acc[r];
      }
    } else {
      acc = wmma_chain<4>(AsX, rB, lo, acc);
#pragma unroll
      for (int r = 0; r < 8; ++r)
        Red[2 * 1024 + r * 128 + nt * 32 + lane] = acc[r];
    }
    __syncthreads();

    // ------- merge partials + fused Euler update (kg0 waves) -------
    if (kg == 0) {
      float* embCur = emb + (size_t)t * (BATCH * HID);
      const int half = lane >> 4, nn = lane & 15;
      const int col = gcol0 + nt * 16 + nn;
#pragma unroll
      for (int r = 0; r < 8; ++r) {
        const int ri = r * 128 + nt * 32 + lane;
        float d = acc[r] + Red[2 * 1024 + ri];
        if (j > 0) d += Red[0 * 1024 + ri];
        if (j < 3) d += Red[1 * 1024 + ri];
        int brow = r + half * 8;        // C-layout: VGPR r holds M=r / M=r+8
        float sp = (t > 0) ? embPrev[(size_t)brow * HID + col] : 0.f;
        embCur[(size_t)brow * HID + col] = 0.99f * sp + 0.01f * d;
      }
    }
    __syncthreads();                    // LDS safe before next staging

    // ------- device-wide step barrier (monotonic counter) -------
    __threadfence();
    if (tid == 0)
      __hip_atomic_fetch_add(counter, 1u, __ATOMIC_RELEASE,
                             __HIP_MEMORY_SCOPE_AGENT);
    if (lane == 0) {                    // every wave acquires (cache inv)
      const unsigned target = (unsigned)NWG * (unsigned)(t + 1);
      while (__hip_atomic_load(counter, __ATOMIC_ACQUIRE,
                               __HIP_MEMORY_SCOPE_AGENT) < target)
        __builtin_amdgcn_s_sleep(1);
    }
    __syncthreads();
  }
}

// ---------------------------------------------------------------------------
// Output projection: out = embeddings @ out_mat.
// One block per M-tile (2048 blocks x 256 threads): the [16 x 1024] A panel
// is staged once into LDS (embeddings read exactly once), then 8 waves each
// run two independent 32-deep WMMA chains (2 N-tiles per wave).
// ---------------------------------------------------------------------------
__global__ void __launch_bounds__(256) out_proj_kernel(
    const float* __restrict__ emb, const _Float16* __restrict__ w,
    float* __restrict__ out) {
  __shared__ _Float16 AsO[32 * TILE_H];   // 32 K-tiles, A-layout (32 KB)

  const int tid = threadIdx.x;
  const int lane = tid & 31;
  const int warp = tid >> 5;
  const int m = blockIdx.x;               // M-tile of [T*B, H]
  const int lo = lane << 4;

  // stage A panel: 16 rows x 1024 cols, f32 -> f16, A-layout
  for (int e = tid; e < 16 * HID; e += 256) {
    int b = e >> 10, K = e & 1023;
    int kt = K >> 5, half, idx;
    inv_k(K & 31, half, idx);
    AsO[kt * TILE_H + (((half << 4) | b) << 4) + idx] =
        (_Float16)emb[(size_t)(m * 16 + b) * HID + K];
  }
  __syncthreads();

  const int n0 = warp * 2, n1 = n0 + 1;   // two N-tiles per wave
  v8f acc0 = {}, acc1 = {};
#pragma unroll
  for (int k = 0; k < 32; ++k) {
    v16h a  = *(const v16h*)&AsO[k * TILE_H + lo];
    v16h b0 = *(const v16h*)&w[OFF_WOUT + (size_t)(n0 * 32 + k) * TILE_H + lo];
    v16h b1 = *(const v16h*)&w[OFF_WOUT + (size_t)(n1 * 32 + k) * TILE_H + lo];
    acc0 = WMMA(a, b0, acc0);
    acc1 = WMMA(a, b1, acc1);
  }
  const int half = lane >> 4, nn = lane & 15;
#pragma unroll
  for (int r = 0; r < 8; ++r) {
    size_t row = (size_t)(m * 16 + r + half * 8) * OUT_DIM;
    out[row + n0 * 16 + nn] = acc0[r];
    out[row + n1 * 16 + nn] = acc1[r];
  }
}

// ---------------------------------------------------------------------------
extern "C" void kernel_launch(void* const* d_in, const int* in_sizes, int n_in,
                              void* d_out, int out_size, void* d_ws,
                              size_t ws_size, hipStream_t stream) {
  (void)in_sizes; (void)n_in; (void)out_size; (void)ws_size;
  const float* x         = (const float*)d_in[0];
  const float* blocks    = (const float*)d_in[1];
  const float* couplings = (const float*)d_in[2];
  // d_in[3] (couple_mask) unused: mask*couplings == couplings elementwise.
  const float* input_mat = (const float*)d_in[4];
  const float* out_mat   = (const float*)d_in[5];

  float* out = (float*)d_out;                                  // [T,B,OUT]
  float* embeddings = out + (size_t)T_STEPS * BATCH * OUT_DIM; // [T,B,H]

  unsigned* counter = (unsigned*)d_ws;
  _Float16* w       = (_Float16*)((char*)d_ws + WS_CTR_BYTES);

  hipMemsetAsync(d_ws, 0, WS_CTR_BYTES, stream);   // reset barrier counter
  swizzle_weights_kernel<<<SZ_TOTAL / 256, 256, 0, stream>>>(
      blocks, couplings, input_mat, out_mat, w);
  rnn_scan_kernel<<<NWG, SCAN_THREADS, 0, stream>>>(x, w, embeddings, counter);
  out_proj_kernel<<<T_STEPS * BATCH / 16, 256, 0, stream>>>(embeddings, w, out);
}
